// SparseMoELayer_33921651704687
// MI455X (gfx1250) — matrix operand order repaired and checked by reading, not compiled, not run on hardware
//
#include <hip/hip_runtime.h>
#include <hip/hip_bf16.h>
#include <stdint.h>

#define NTOK   8192      // B*S
#define DDIM   2048
#define HDIM   2048
#define ODIM   2048
#define NEXP   8

#define ASTR   40        // LDS row stride in ushorts: 80B, 16B-aligned, bank-spread

#if defined(__has_builtin)
#if __has_builtin(__builtin_amdgcn_global_load_async_to_lds_b128)
#define USE_ASYNC_LDS 1
#endif
#endif

typedef __attribute__((ext_vector_type(16))) __bf16 v16bf;
typedef __attribute__((ext_vector_type(8)))  float  v8f;
typedef int v4i_ __attribute__((vector_size(16)));   // matches builtin param type

union Frag16 { uint4 u4[2]; v16bf v; };

__device__ __forceinline__ unsigned short f2bf(float f) {
  union { float f; unsigned u; } c; c.f = f;
  unsigned r = c.u + 0x7FFFu + ((c.u >> 16) & 1u);   // round-to-nearest-even
  return (unsigned short)(r >> 16);
}

__device__ __forceinline__ float gelu_exact(float x) {
  return 0.5f * x * (1.0f + erff(x * 0.70710678118654752f));
}

#ifdef USE_ASYNC_LDS
// AS1 value == generic value; AS3 value == low 32 bits of generic LDS address.
__device__ __forceinline__ void async_ld_b128(const unsigned short* g,
                                              unsigned short* l) {
  __builtin_amdgcn_global_load_async_to_lds_b128(
      (__attribute__((address_space(1))) v4i_*)(uintptr_t)g,
      (__attribute__((address_space(3))) v4i_*)(uint32_t)(uintptr_t)l,
      0, 0);
}
#endif

// ---------------------------------------------------------------- utilities
__global__ void zero_router_state(int* cnt, float* psum) {
  int t = threadIdx.x;
  if (t < NEXP) { cnt[t] = 0; psum[t] = 0.f; }
}

__global__ void cvt_f32_bf16(const float* __restrict__ src,
                             unsigned short* __restrict__ dst, long n) {
  long i = (long)blockIdx.x * blockDim.x + threadIdx.x;
  long stride = (long)gridDim.x * blockDim.x;
  for (; i < n; i += stride) dst[i] = f2bf(src[i]);
}

// ---------------------------------------------------------------- router
__global__ void router_kernel(const float* __restrict__ x,
                              const float* __restrict__ rw,
                              const float* __restrict__ rb,
                              int* __restrict__ cnt, float* __restrict__ psum,
                              int* __restrict__ idx, float* __restrict__ wgt) {
  int t = blockIdx.x * blockDim.x + threadIdx.x;
  if (t >= NTOK) return;
  float acc[NEXP];
  #pragma unroll
  for (int e = 0; e < NEXP; ++e) acc[e] = rb[e];
  const float4* xv = (const float4*)(x + (size_t)t * DDIM);
  for (int d4 = 0; d4 < DDIM / 4; ++d4) {
    float4 xq = xv[d4];
    #pragma unroll
    for (int e = 0; e < NEXP; ++e) {
      float4 wq = *(const float4*)(rw + (size_t)e * DDIM + d4 * 4);
      acc[e] += xq.x * wq.x + xq.y * wq.y + xq.z * wq.z + xq.w * wq.w;
    }
  }
  float m = acc[0];
  #pragma unroll
  for (int e = 1; e < NEXP; ++e) m = fmaxf(m, acc[e]);
  float p[NEXP]; float s = 0.f;
  #pragma unroll
  for (int e = 0; e < NEXP; ++e) { p[e] = __expf(acc[e] - m); s += p[e]; }
  float inv = 1.f / s;
  #pragma unroll
  for (int e = 0; e < NEXP; ++e) atomicAdd(&psum[e], p[e] * inv);
  int i1 = 0;
  #pragma unroll
  for (int e = 1; e < NEXP; ++e) if (p[e] > p[i1]) i1 = e;
  int i2 = (i1 == 0) ? 1 : 0;
  #pragma unroll
  for (int e = 0; e < NEXP; ++e) if (e != i1 && p[e] > p[i2]) i2 = e;
  float denom = p[i1] + p[i2];
  float w1v = p[i1] / denom, w2v = p[i2] / denom;
  int pos1 = atomicAdd(&cnt[i1], 1);
  idx[i1 * NTOK + pos1] = t; wgt[i1 * NTOK + pos1] = w1v;
  int pos2 = atomicAdd(&cnt[i2], 1);
  idx[i2 * NTOK + pos2] = t; wgt[i2 * NTOK + pos2] = w2v;
}

__global__ void aux_kernel(const float* __restrict__ psum, float* out_aux) {
  if (threadIdx.x == 0 && blockIdx.x == 0) {
    float a = 0.f;
    for (int e = 0; e < NEXP; ++e) {
      float u = psum[e] / (float)NTOK;
      a += u * u;
    }
    out_aux[0] = a * (float)NEXP * 0.001f;
  }
}

// ---------------------------------------------------------------- GEMM1
// hid[pos, h] = gelu(x@w1^T + b1) * gelu(x@w3^T + b3), rows gathered by idxList
// Tile: 128 (M) x 128 (N) per WG; 8 waves, wave w owns cols w*16..w*16+15,
// 8 M-subtiles per wave. A tile double-buffered in LDS via async loads.
// A fragments software-pipelined depth-1 (load s+1 before WMMAs of s).
__global__ __launch_bounds__(256)
void moe_gemm1(const unsigned short* __restrict__ A,     // [NTOK, K] bf16
               const unsigned short* __restrict__ B1,    // [HDIM, K] bf16 (w1)
               const unsigned short* __restrict__ B3,    // [HDIM, K] bf16 (w3)
               const float* __restrict__ bias1,
               const float* __restrict__ bias3,
               unsigned short* __restrict__ hid,         // [NTOK, HDIM] bf16
               const int* __restrict__ idxList,          // null => identity
               const int* __restrict__ countPtr,         // null => NTOK
               int K) {
  const int count = countPtr ? *countPtr : NTOK;
  const int m0 = blockIdx.y * 128;
  if (m0 >= count) return;
  const int n0 = blockIdx.x * 128;
  const int lane = threadIdx.x & 31;
  const int wave = threadIdx.x >> 5;

  __shared__ alignas(16) unsigned short As[2][128 * ASTR];

  // Fill: thread t handles rows (t>>2) and (t>>2)+64, K-segment (t&3)*8
  const int frow = threadIdx.x >> 2;
  const int fseg = (threadIdx.x & 3) * 8;
  int p0 = m0 + frow, p1 = m0 + frow + 64;
  int g0 = idxList ? ((p0 < count) ? idxList[p0] : idxList[0]) : p0;
  int g1 = idxList ? ((p1 < count) ? idxList[p1] : idxList[0]) : p1;
  const unsigned short* arow0 = A + (size_t)g0 * K + fseg;
  const unsigned short* arow1 = A + (size_t)g1 * K + fseg;
  unsigned short* l0 = &As[0][frow * ASTR + fseg];
  unsigned short* l1 = &As[0][(frow + 64) * ASTR + fseg];
  const int lbuf = 128 * ASTR;  // ushorts per buffer

  // B frags: lane n<16 -> col n0+w*16+n, K 0..7,16..23; lane>=16 same col, K 8..15,24..31
  const int col = n0 + wave * 16 + (lane & 15);
  const unsigned short* b1p = B1 + (size_t)col * K + ((lane >> 4) << 3);
  const unsigned short* b3p = B3 + (size_t)col * K + ((lane >> 4) << 3);
  const int kb = (lane >> 4) << 3;

  v8f acc1[8] = {};
  v8f acc3[8] = {};

  const int steps = K / 32;

  // prologue: tile 0 -> buffer 0
#ifdef USE_ASYNC_LDS
  async_ld_b128(arow0, l0);
  async_ld_b128(arow1, l1);
  asm volatile("s_wait_asynccnt 0" ::: "memory");
#else
  *(uint4*)l0 = *(const uint4*)arow0;
  *(uint4*)l1 = *(const uint4*)arow1;
#endif
  __syncthreads();

  for (int i = 0; i < steps; ++i) {
    const int cur = i & 1;
    const int kk = i * 32;

    // prefetch next A tile into the other buffer (overlaps with WMMA below)
#ifdef USE_ASYNC_LDS
    if (i + 1 < steps) {
      async_ld_b128(arow0 + kk + 32, l0 + (1 - cur) * lbuf);
      async_ld_b128(arow1 + kk + 32, l1 + (1 - cur) * lbuf);
    }
#else
    uint4 pr0, pr1;
    if (i + 1 < steps) {
      pr0 = *(const uint4*)(arow0 + kk + 32);
      pr1 = *(const uint4*)(arow1 + kk + 32);
    }
#endif

    Frag16 fb1, fb3;
    fb1.u4[0] = *(const uint4*)(b1p + kk);
    fb1.u4[1] = *(const uint4*)(b1p + kk + 16);
    fb3.u4[0] = *(const uint4*)(b3p + kk);
    fb3.u4[1] = *(const uint4*)(b3p + kk + 16);

    const unsigned short* buf = &As[cur][0];

    // depth-1 pipelined A fragments: load s+1 while WMMAs of s execute
    Frag16 fa_cur, fa_nxt;
    {
      const unsigned short* ar = buf + (lane & 15) * ASTR + kb;
      fa_cur.u4[0] = *(const uint4*)(ar);
      fa_cur.u4[1] = *(const uint4*)(ar + 16);
    }
    #pragma unroll
    for (int s = 0; s < 8; ++s) {
      if (s + 1 < 8) {
        const unsigned short* ar = buf + ((s + 1) * 16 + (lane & 15)) * ASTR + kb;
        fa_nxt.u4[0] = *(const uint4*)(ar);
        fa_nxt.u4[1] = *(const uint4*)(ar + 16);
      }
      acc1[s] = __builtin_amdgcn_wmma_f32_16x16x32_bf16(
          false, fa_cur.v, false, fb1.v, (short)0, acc1[s], false, false);
      acc3[s] = __builtin_amdgcn_wmma_f32_16x16x32_bf16(
          false, fa_cur.v, false, fb3.v, (short)0, acc3[s], false, false);
      fa_cur = fa_nxt;   // pure renaming under full unroll
    }

#ifdef USE_ASYNC_LDS
    asm volatile("s_wait_asynccnt 0" ::: "memory");
#else
    if (i + 1 < steps) {
      *(uint4*)(l0 + (1 - cur) * lbuf) = pr0;
      *(uint4*)(l1 + (1 - cur) * lbuf) = pr1;
    }
#endif
    __syncthreads();
  }

  // C layout: lane&15 -> N col, (lane>>4)*8 + vgpr -> M row
  const float bb1 = bias1[col];
  const float bb3 = bias3[col];
  const int mbase = m0 + ((lane >> 4) << 3);
  #pragma unroll
  for (int s = 0; s < 8; ++s) {
    #pragma unroll
    for (int r = 0; r < 8; ++r) {
      int row = mbase + s * 16 + r;
      float h1 = acc1[s][r] + bb1;
      float h3 = acc3[s][r] + bb3;
      hid[(size_t)row * HDIM + col] = f2bf(gelu_exact(h1) * gelu_exact(h3));
    }
  }
}

// ---------------------------------------------------------------- GEMM2
// out[token, o] (+)= wgt[pos] * (hid@w2^T + b2); same tiling as GEMM1.
// All 8 A fragments preloaded (64 VGPRs) for maximal ds/WMMA overlap.
template <bool ACCUM>
__global__ __launch_bounds__(256)
void moe_gemm2(const unsigned short* __restrict__ A,     // hid [NTOK, K] bf16
               const unsigned short* __restrict__ B,     // w2 [ODIM, K] bf16
               const float* __restrict__ bias,           // [ODIM]
               float* __restrict__ out,                  // [NTOK, ODIM] f32
               const int* __restrict__ idxList,          // null => identity
               const float* __restrict__ wgtList,        // null => 1.0
               const int* __restrict__ countPtr,         // null => NTOK
               int K) {
  const int count = countPtr ? *countPtr : NTOK;
  const int m0 = blockIdx.y * 128;
  if (m0 >= count) return;
  const int n0 = blockIdx.x * 128;
  const int lane = threadIdx.x & 31;
  const int wave = threadIdx.x >> 5;

  __shared__ alignas(16) unsigned short As[2][128 * ASTR];

  const int frow = threadIdx.x >> 2;
  const int fseg = (threadIdx.x & 3) * 8;
  const unsigned short* arow0 = A + (size_t)(m0 + frow) * K + fseg;
  const unsigned short* arow1 = A + (size_t)(m0 + frow + 64) * K + fseg;
  unsigned short* l0 = &As[0][frow * ASTR + fseg];
  unsigned short* l1 = &As[0][(frow + 64) * ASTR + fseg];
  const int lbuf = 128 * ASTR;

  const int col = n0 + wave * 16 + (lane & 15);
  const unsigned short* bp = B + (size_t)col * K + ((lane >> 4) << 3);
  const int kb = (lane >> 4) << 3;

  v8f acc[8] = {};
  const int steps = K / 32;

#ifdef USE_ASYNC_LDS
  async_ld_b128(arow0, l0);
  async_ld_b128(arow1, l1);
  asm volatile("s_wait_asynccnt 0" ::: "memory");
#else
  *(uint4*)l0 = *(const uint4*)arow0;
  *(uint4*)l1 = *(const uint4*)arow1;
#endif
  __syncthreads();

  for (int i = 0; i < steps; ++i) {
    const int cur = i & 1;
    const int kk = i * 32;

#ifdef USE_ASYNC_LDS
    if (i + 1 < steps) {
      async_ld_b128(arow0 + kk + 32, l0 + (1 - cur) * lbuf);
      async_ld_b128(arow1 + kk + 32, l1 + (1 - cur) * lbuf);
    }
#else
    uint4 pr0, pr1;
    if (i + 1 < steps) {
      pr0 = *(const uint4*)(arow0 + kk + 32);
      pr1 = *(const uint4*)(arow1 + kk + 32);
    }
#endif

    Frag16 fb;
    fb.u4[0] = *(const uint4*)(bp + kk);
    fb.u4[1] = *(const uint4*)(bp + kk + 16);

    const unsigned short* buf = &As[cur][0];

    // preload all A fragments; scheduler interleaves dscnt waits with WMMAs
    Frag16 fa[8];
    #pragma unroll
    for (int s = 0; s < 8; ++s) {
      const unsigned short* ar = buf + (s * 16 + (lane & 15)) * ASTR + kb;
      fa[s].u4[0] = *(const uint4*)(ar);
      fa[s].u4[1] = *(const uint4*)(ar + 16);
    }
    #pragma unroll
    for (int s = 0; s < 8; ++s) {
      acc[s] = __builtin_amdgcn_wmma_f32_16x16x32_bf16(
          false, fa[s].v, false, fb.v, (short)0, acc[s], false, false);
    }

#ifdef USE_ASYNC_LDS
    asm volatile("s_wait_asynccnt 0" ::: "memory");
#else
    if (i + 1 < steps) {
      *(uint4*)(l0 + (1 - cur) * lbuf) = pr0;
      *(uint4*)(l1 + (1 - cur) * lbuf) = pr1;
    }
#endif
    __syncthreads();
  }

  const float bb = bias[col];
  const int mbase = ((lane >> 4) << 3);
  #pragma unroll
  for (int s = 0; s < 8; ++s) {
    #pragma unroll
    for (int r = 0; r < 8; ++r) {
      int pos = m0 + mbase + s * 16 + r;
      if (pos < count) {
        int token = idxList ? idxList[pos] : pos;
        float scale = wgtList ? wgtList[pos] : 1.0f;
        float val = scale * (acc[s][r] + bb);
        size_t o = (size_t)token * ODIM + col;
        if (ACCUM) out[o] += val;   // experts run sequentially: no races
        else       out[o]  = val;   // shared expert initializes out fully
      }
    }
  }
}

// ---------------------------------------------------------------- launch
extern "C" void kernel_launch(void* const* d_in, const int* in_sizes, int n_in,
                              void* d_out, int out_size, void* d_ws, size_t ws_size,
                              hipStream_t stream) {
  (void)in_sizes; (void)n_in; (void)out_size; (void)ws_size;
  const float* x   = (const float*)d_in[0];
  const float* rw  = (const float*)d_in[1];
  const float* rb  = (const float*)d_in[2];
  const float* w1  = (const float*)d_in[3];
  const float* b1  = (const float*)d_in[4];
  const float* w2  = (const float*)d_in[5];
  const float* b2  = (const float*)d_in[6];
  const float* w3  = (const float*)d_in[7];
  const float* b3  = (const float*)d_in[8];
  const float* sw1 = (const float*)d_in[9];
  const float* sb1 = (const float*)d_in[10];
  const float* sw2 = (const float*)d_in[11];
  const float* sb2 = (const float*)d_in[12];
  const float* sw3 = (const float*)d_in[13];
  const float* sb3 = (const float*)d_in[14];
  float* out = (float*)d_out;

  char* wsb = (char*)d_ws;
  size_t off = 0;
  auto alloc = [&](size_t bytes) -> char* {
    char* p = wsb + off;
    off = (off + bytes + 255) & ~(size_t)255;
    return p;
  };
  unsigned short* xb   = (unsigned short*)alloc((size_t)NTOK * DDIM * 2);
  unsigned short* wb1  = (unsigned short*)alloc((size_t)NEXP * HDIM * DDIM * 2);
  unsigned short* wb2  = (unsigned short*)alloc((size_t)NEXP * ODIM * HDIM * 2);
  unsigned short* wb3  = (unsigned short*)alloc((size_t)NEXP * HDIM * DDIM * 2);
  unsigned short* swb1 = (unsigned short*)alloc((size_t)HDIM * DDIM * 2);
  unsigned short* swb2 = (unsigned short*)alloc((size_t)ODIM * HDIM * 2);
  unsigned short* swb3 = (unsigned short*)alloc((size_t)HDIM * DDIM * 2);
  unsigned short* hid  = (unsigned short*)alloc((size_t)NTOK * HDIM * 2);
  int*   cnt  = (int*)alloc(NEXP * sizeof(int));
  float* psum = (float*)alloc(NEXP * sizeof(float));
  int*   idx  = (int*)alloc((size_t)NEXP * NTOK * sizeof(int));
  float* wgt  = (float*)alloc((size_t)NEXP * NTOK * sizeof(float));

  dim3 blk(256);

  zero_router_state<<<1, 64, 0, stream>>>(cnt, psum);

  cvt_f32_bf16<<<2048, blk, 0, stream>>>(x, xb, (long)NTOK * DDIM);
  cvt_f32_bf16<<<4096, blk, 0, stream>>>(w1, wb1, (long)NEXP * HDIM * DDIM);
  cvt_f32_bf16<<<4096, blk, 0, stream>>>(w2, wb2, (long)NEXP * ODIM * HDIM);
  cvt_f32_bf16<<<4096, blk, 0, stream>>>(w3, wb3, (long)NEXP * HDIM * DDIM);
  cvt_f32_bf16<<<1024, blk, 0, stream>>>(sw1, swb1, (long)HDIM * DDIM);
  cvt_f32_bf16<<<1024, blk, 0, stream>>>(sw2, swb2, (long)ODIM * HDIM);
  cvt_f32_bf16<<<1024, blk, 0, stream>>>(sw3, swb3, (long)HDIM * DDIM);

  router_kernel<<<NTOK / 256, blk, 0, stream>>>(x, rw, rb, cnt, psum, idx, wgt);
  aux_kernel<<<1, 32, 0, stream>>>(psum, out + (size_t)NTOK * ODIM);

  dim3 g1(HDIM / 128, NTOK / 128);   // (16, 64)
  dim3 g2(ODIM / 128, NTOK / 128);

  // shared expert: fully writes out (also clears poisoned buffer)
  moe_gemm1<<<g1, blk, 0, stream>>>(xb, swb1, swb3, sb1, sb3, hid,
                                    nullptr, nullptr, DDIM);
  moe_gemm2<false><<<g2, blk, 0, stream>>>(hid, swb2, sb2, out,
                                           nullptr, nullptr, nullptr, HDIM);

  // routed experts, sequential: hid buffer reused, out accumulated race-free
  for (int e = 0; e < NEXP; ++e) {
    moe_gemm1<<<g1, blk, 0, stream>>>(
        xb, wb1 + (size_t)e * HDIM * DDIM, wb3 + (size_t)e * HDIM * DDIM,
        b1 + (size_t)e * HDIM, b3 + (size_t)e * HDIM, hid,
        idx + (size_t)e * NTOK, cnt + e, DDIM);
    moe_gemm2<true><<<g2, blk, 0, stream>>>(
        hid, wb2 + (size_t)e * ODIM * HDIM, b2 + (size_t)e * ODIM, out,
        idx + (size_t)e * NTOK, wgt + (size_t)e * NTOK, cnt + e, HDIM);
  }
}
